// WindowAttention_55800215109970
// MI455X (gfx1250) — compile-verified, matrix operand order checked by reading, not run on hardware
//
#include <hip/hip_runtime.h>
#include <hip/hip_bf16.h>

typedef __bf16 bf16;
typedef __bf16 v16bf __attribute__((ext_vector_type(16)));
typedef __bf16 v8bf  __attribute__((ext_vector_type(8)));
typedef float  v8f   __attribute__((ext_vector_type(8)));

#define DIM     512
#define NHEADS  16
#define NTOK    49
#define NPAD    64
#define HD      32
#define B_TOT   2048
#define NW      64
#define M_TOT   (B_TOT * NTOK)          /* 100352, divisible by 128 */
#define QK_SCALE 0.17677669529663687f   /* 32^-0.5 */

__device__ __forceinline__ bf16 f2bf(float f) {
  unsigned u = __builtin_bit_cast(unsigned, f);
  u += 0x7fffu + ((u >> 16) & 1u);            // round-to-nearest-even
  unsigned short h = (unsigned short)(u >> 16);
  return __builtin_bit_cast(bf16, h);
}

__device__ __forceinline__ v16bf cat16(v8bf lo, v8bf hi) {
  return __builtin_shufflevector(lo, hi, 0,1,2,3,4,5,6,7,8,9,10,11,12,13,14,15);
}

// Load one 16x32 bf16 WMMA fragment (A, or B via B^T) from a row-major matrix.
// CDNA5 16-bit A layout: lane l holds row (l&15); half (l>>4) selects
// K sub-ranges [half*8, half*8+8) (VGPR0-3) and [16+half*8, ...) (VGPR4-7).
__device__ __forceinline__ v16bf load_frag(const bf16* base, int rowbase,
                                           int stride, int lane, int kofs) {
  int r = rowbase + (lane & 15);
  int half = lane >> 4;
  const bf16* p = base + (size_t)r * stride + kofs + half * 8;
  v8bf lo = *(const v8bf*)p;
  v8bf hi = *(const v8bf*)(p + 16);
  return cat16(lo, hi);
}

#define WMMA_BF16(A, B, C) \
  __builtin_amdgcn_wmma_f32_16x16x32_bf16(false, (A), false, (B), (short)0, (C), false, false)

// CDNA5 async global->LDS copy (ASYNCcnt-tracked, no VGPR round trip).
__device__ __forceinline__ void async_copy16(const void* g, void* l) {
  unsigned lds = (unsigned)(uintptr_t)l;                 // low 32b of flat = LDS offset
  unsigned long long ga = (unsigned long long)(uintptr_t)g;
  asm volatile("global_load_async_to_lds_b128 %0, %1, off"
               :: "v"(lds), "v"(ga) : "memory");
}
__device__ __forceinline__ void wait_async0() {
  asm volatile("s_wait_asynccnt 0" ::: "memory");
}

// ----------------------------------------------------------- x f32 -> bf16
// One pass so the GEMM re-reads (12 column tiles) hit a 103 MB bf16 copy that
// fits the 192 MB L2, and the GEMM hot loop carries no conversion VALU.
__global__ __launch_bounds__(256)
void convert_x_kernel(const float* __restrict__ X, bf16* __restrict__ XB) {
  size_t c = (size_t)blockIdx.x * 256 + threadIdx.x;   // one 8-elem chunk each
  const float* src = X + c * 8;
  v8bf o;
#pragma unroll
  for (int e = 0; e < 8; ++e) o[e] = f2bf(src[e]);
  *(v8bf*)(XB + c * 8) = o;
}

// ---------------------------------------------------------------- weights cvt
__global__ __launch_bounds__(256)
void convert_w_kernel(const float* __restrict__ qkvw, const float* __restrict__ projw,
                      bf16* __restrict__ w1, bf16* __restrict__ w2) {
  int i = blockIdx.x * 256 + threadIdx.x;
  if (i < 3 * DIM * DIM) w1[i] = f2bf(qkvw[i]);
  if (i < DIM * DIM)     w2[i] = f2bf(projw[i]);
}

// ------------------------------------------------- zero pad rows of q/k/v^T
__global__ __launch_bounds__(256)
void pad_zero_kernel(bf16* __restrict__ Q, bf16* __restrict__ K, bf16* __restrict__ VT) {
  int i = blockIdx.x * 256 + threadIdx.x;
  if (i >= B_TOT * NHEADS * 15 * 32) return;
  int bh  = i / 480;
  int rem = i - bh * 480;
  int nr  = NTOK + rem / 32;
  int dd  = rem & 31;
  bf16 z = __builtin_bit_cast(bf16, (unsigned short)0);
  size_t o = (size_t)bh * (NPAD * HD);
  Q [o + nr * HD + dd] = z;
  K [o + nr * HD + dd] = z;
  VT[o + dd * NPAD + nr] = z;
}

// ------------------------------------------------------------------- GEMM 1
// qkv = x @ qkv_w^T + qkv_b, scattered into q (scaled), k, v^T padded tiles.
// 128x128 block tile, 8 waves of 32x64, double-buffered LDS, pure async fills.
#define BM 128
#define BN 128
#define BK 32
#define LDA 40
#define LDB 40

__global__ __launch_bounds__(256)
void gemm_qkv_kernel(const bf16* __restrict__ X, const bf16* __restrict__ W,
                     const float* __restrict__ bias,
                     bf16* __restrict__ Q, bf16* __restrict__ K, bf16* __restrict__ VT) {
  __shared__ __align__(16) bf16 sA[2][BM * LDA];
  __shared__ __align__(16) bf16 sB[2][BN * LDB];
  const int tid  = threadIdx.x;
  const int lane = tid & 31;
  const int wave = tid >> 5;
  const int wm = (wave & 3) * 32;
  const int wn = (wave >> 2) * 64;
  const int N0 = blockIdx.x * BN;
  const int M0 = blockIdx.y * BM;

  v8f acc[2][4] = {};

  auto fill = [&](int kt, int buf) {
    int k0 = kt * BK;
    for (int c = tid; c < BM * 4; c += 256) {
      int r = c >> 2, ch = (c & 3) * 8;
      async_copy16(X + (size_t)(M0 + r) * DIM + k0 + ch, &sA[buf][r * LDA + ch]);
    }
    for (int c = tid; c < BN * 4; c += 256) {
      int r = c >> 2, ch = (c & 3) * 8;
      async_copy16(W + (size_t)(N0 + r) * DIM + k0 + ch, &sB[buf][r * LDB + ch]);
    }
  };

  fill(0, 0);
  const int KT = DIM / BK;   // 16
#pragma unroll 2
  for (int kt = 0; kt < KT; ++kt) {
    int cur = kt & 1;
    wait_async0();
    __syncthreads();
    if (kt + 1 < KT) fill(kt + 1, cur ^ 1);

    v16bf a0 = load_frag(sA[cur], wm,      LDA, lane, 0);
    v16bf a1 = load_frag(sA[cur], wm + 16, LDA, lane, 0);
#pragma unroll
    for (int ni = 0; ni < 4; ++ni) {
      v16bf b = load_frag(sB[cur], wn + ni * 16, LDB, lane, 0);
      acc[0][ni] = WMMA_BF16(a0, b, acc[0][ni]);
      acc[1][ni] = WMMA_BF16(a1, b, acc[1][ni]);
    }
  }

  // Epilogue: add bias, scatter to q/k/v^T padded per-(b,h) tiles.
  const int nl = lane & 15;
  const int mh = (lane >> 4) * 8;
#pragma unroll
  for (int ni = 0; ni < 4; ++ni) {
    int ncol = N0 + wn + ni * 16 + nl;
    float bv = bias[ncol];
    int which = ncol >> 9;
    int h  = (ncol >> 5) & 15;
    int dd = ncol & 31;
#pragma unroll
    for (int mi = 0; mi < 2; ++mi) {
      v8f a = acc[mi][ni];
#pragma unroll
      for (int r = 0; r < 8; ++r) {
        int m  = M0 + wm + mi * 16 + mh + r;
        int b  = m / NTOK;
        int nr = m - b * NTOK;
        size_t bho = (size_t)(b * NHEADS + h) * (NPAD * HD);
        float val = a[r] + bv;
        if (which == 0)      Q [bho + nr * HD + dd] = f2bf(val * QK_SCALE);
        else if (which == 1) K [bho + nr * HD + dd] = f2bf(val);
        else                 VT[bho + dd * NPAD + nr] = f2bf(val);
      }
    }
  }
}

// ---------------------------------------------------------------- attention
#define SSTR 68
#define PSTR 72

__global__ __launch_bounds__(128)
void attn_kernel(const bf16* __restrict__ Q, const bf16* __restrict__ K,
                 const bf16* __restrict__ VT, const float* __restrict__ mask,
                 const float* __restrict__ rpb, bf16* __restrict__ CTX) {
  __shared__ __align__(16) float sS[64 * SSTR];
  __shared__ __align__(16) bf16  sP[64 * PSTR];
  __shared__ __align__(16) float sBM[NTOK * NTOK];

  const int bh = blockIdx.x;
  const int b  = bh >> 4;
  const int h  = bh & 15;
  const int w  = b & (NW - 1);
  const int tid  = threadIdx.x;
  const int lane = tid & 31;
  const int wave = tid >> 5;

  for (int idx = tid; idx < NTOK * NTOK; idx += 128) {
    int m = idx / NTOK, n = idx - m * NTOK;
    int mi = m / 7, mj = m - mi * 7;
    int ni = n / 7, nj = n - ni * 7;
    int rel = (mi - ni + 6) * 13 + (mj - nj + 6);
    sBM[idx] = rpb[rel * NHEADS + h] + mask[((size_t)w * NTOK + m) * NTOK + n];
  }

  const bf16* qp = Q  + (size_t)bh * (NPAD * HD);
  const bf16* kp = K  + (size_t)bh * (NPAD * HD);
  const bf16* vp = VT + (size_t)bh * (NPAD * HD);

  v16bf aq = load_frag(qp, wave * 16, HD, lane, 0);
#pragma unroll
  for (int nt = 0; nt < 4; ++nt) {
    v16bf bk = load_frag(kp, nt * 16, HD, lane, 0);
    v8f c = {};
    c = WMMA_BF16(aq, bk, c);
    int nc = nt * 16 + (lane & 15);
    int mb = wave * 16 + (lane >> 4) * 8;
#pragma unroll
    for (int r = 0; r < 8; ++r) sS[(mb + r) * SSTR + nc] = c[r];
  }
  __syncthreads();   // sBM ready (S tiles are per-wave private)

  int row = wave * 16 + (lane & 15);
  int rbm = (row < NTOK) ? row : 0;
  int ch  = (lane >> 4) * 32;
  float mx = -3.0e38f;
#pragma unroll
  for (int jj = 0; jj < 32; ++jj) {
    int j = ch + jj;
    if (j < NTOK) {
      float v = sS[row * SSTR + j] + sBM[rbm * NTOK + j];
      sS[row * SSTR + j] = v;
      mx = fmaxf(mx, v);
    }
  }
  mx = fmaxf(mx, __shfl_xor(mx, 16, 32));
  float sum = 0.f;
#pragma unroll
  for (int jj = 0; jj < 32; ++jj) {
    int j = ch + jj;
    if (j < NTOK) {
      float e = __expf(sS[row * SSTR + j] - mx);
      sS[row * SSTR + j] = e;
      sum += e;
    }
  }
  sum += __shfl_xor(sum, 16, 32);
  float inv = 1.f / sum;
  bf16 z = __builtin_bit_cast(bf16, (unsigned short)0);
#pragma unroll
  for (int jj = 0; jj < 32; ++jj) {
    int j = ch + jj;
    sP[row * PSTR + j] = (j < NTOK) ? f2bf(sS[row * SSTR + j] * inv) : z;
  }

#pragma unroll
  for (int nt = 0; nt < 2; ++nt) {
    v8f c = {};
#pragma unroll
    for (int kc = 0; kc < 2; ++kc) {
      v16bf pa = load_frag(sP, wave * 16, PSTR, lane, kc * 32);
      v16bf vb = load_frag(vp, nt * 16, NPAD, lane, kc * 32);
      c = WMMA_BF16(pa, vb, c);
    }
    int ncol = h * HD + nt * 16 + (lane & 15);
    int mb = wave * 16 + (lane >> 4) * 8;
#pragma unroll
    for (int r = 0; r < 8; ++r) {
      int m = mb + r;
      if (m < NTOK)
        CTX[((size_t)b * NTOK + m) * DIM + ncol] = f2bf(c[r]);
    }
  }
}

// ------------------------------------------------------------------- GEMM 2
__global__ __launch_bounds__(256)
void gemm_proj_kernel(const bf16* __restrict__ A, const bf16* __restrict__ W,
                      const float* __restrict__ bias, float* __restrict__ out) {
  __shared__ __align__(16) bf16 sA[2][BM * LDA];
  __shared__ __align__(16) bf16 sB[2][BN * LDB];
  const int tid  = threadIdx.x;
  const int lane = tid & 31;
  const int wave = tid >> 5;
  const int wm = (wave & 3) * 32;
  const int wn = (wave >> 2) * 64;
  const int N0 = blockIdx.x * BN;
  const int M0 = blockIdx.y * BM;

  v8f acc[2][4] = {};

  auto fill = [&](int kt, int buf) {
    int k0 = kt * BK;
    for (int c = tid; c < BM * 4; c += 256) {
      int r = c >> 2, ch = (c & 3) * 8;
      async_copy16(A + (size_t)(M0 + r) * DIM + k0 + ch, &sA[buf][r * LDA + ch]);
    }
    for (int c = tid; c < BN * 4; c += 256) {
      int r = c >> 2, ch = (c & 3) * 8;
      async_copy16(W + (size_t)(N0 + r) * DIM + k0 + ch, &sB[buf][r * LDB + ch]);
    }
  };

  fill(0, 0);
  const int KT = DIM / BK;
#pragma unroll 2
  for (int kt = 0; kt < KT; ++kt) {
    int cur = kt & 1;
    wait_async0();
    __syncthreads();
    if (kt + 1 < KT) fill(kt + 1, cur ^ 1);

    v16bf a0 = load_frag(sA[cur], wm,      LDA, lane, 0);
    v16bf a1 = load_frag(sA[cur], wm + 16, LDA, lane, 0);
#pragma unroll
    for (int ni = 0; ni < 4; ++ni) {
      v16bf b = load_frag(sB[cur], wn + ni * 16, LDB, lane, 0);
      acc[0][ni] = WMMA_BF16(a0, b, acc[0][ni]);
      acc[1][ni] = WMMA_BF16(a1, b, acc[1][ni]);
    }
  }

  const int nl = lane & 15;
  const int mh = (lane >> 4) * 8;
#pragma unroll
  for (int ni = 0; ni < 4; ++ni) {
    int ncol = N0 + wn + ni * 16 + nl;
    float bv = bias[ncol];
#pragma unroll
    for (int mi = 0; mi < 2; ++mi) {
      v8f a = acc[mi][ni];
#pragma unroll
      for (int r = 0; r < 8; ++r) {
        int m = M0 + wm + mi * 16 + mh + r;
        out[(size_t)m * DIM + ncol] = a[r] + bv;
      }
    }
  }
}

// ------------------------------------------------------------------- launch
extern "C" void kernel_launch(void* const* d_in, const int* in_sizes, int n_in,
                              void* d_out, int out_size, void* d_ws, size_t ws_size,
                              hipStream_t stream) {
  (void)in_sizes; (void)n_in; (void)out_size; (void)ws_size;
  const float* x      = (const float*)d_in[0];
  const float* mask   = (const float*)d_in[1];
  const float* qkv_w  = (const float*)d_in[2];
  const float* qkv_b  = (const float*)d_in[3];
  const float* proj_w = (const float*)d_in[4];
  const float* proj_b = (const float*)d_in[5];
  const float* rpb    = (const float*)d_in[6];
  float* out = (float*)d_out;

  char* ws = (char*)d_ws;
  bf16* w1  = (bf16*)ws; ws += (size_t)3 * DIM * DIM * 2;
  bf16* w2  = (bf16*)ws; ws += (size_t)DIM * DIM * 2;
  bf16* xb  = (bf16*)ws; ws += (size_t)M_TOT * DIM * 2;
  bf16* q   = (bf16*)ws; ws += (size_t)B_TOT * NHEADS * NPAD * HD * 2;
  bf16* k   = (bf16*)ws; ws += (size_t)B_TOT * NHEADS * NPAD * HD * 2;
  bf16* vt  = (bf16*)ws; ws += (size_t)B_TOT * NHEADS * NPAD * HD * 2;
  bf16* ctx = (bf16*)ws; ws += (size_t)M_TOT * DIM * 2;

  convert_x_kernel<<<(M_TOT * DIM / 8) / 256, 256, 0, stream>>>(x, xb);
  convert_w_kernel<<<(3 * DIM * DIM + 255) / 256, 256, 0, stream>>>(qkv_w, proj_w, w1, w2);
  pad_zero_kernel<<<(B_TOT * NHEADS * 15 * 32 + 255) / 256, 256, 0, stream>>>(q, k, vt);
  gemm_qkv_kernel<<<dim3(3 * DIM / BN, M_TOT / BM), 256, 0, stream>>>(xb, w1, qkv_b, q, k, vt);
  attn_kernel<<<B_TOT * NHEADS, 128, 0, stream>>>(q, k, vt, mask, rpb, ctx);
  gemm_proj_kernel<<<dim3(DIM / BN, M_TOT / BM), 256, 0, stream>>>(ctx, w2, proj_b, out);
}
